// ECCPagedAttentionShim_17179869349
// MI455X (gfx1250) — matrix operand order repaired and checked by reading, not compiled, not run on hardware
//
#include <hip/hip_runtime.h>
#include <stdint.h>

// ---------------------------------------------------------------------------
// Problem constants (from the reference)
// ---------------------------------------------------------------------------
#define NUM_HEADS    32
#define NUM_KV_HEADS 8
#define HEAD_DIM     128
#define HIDDEN       4096
#define KV_HIDDEN    1024
#define BATCH        2
#define SEQ          2048

typedef __attribute__((ext_vector_type(16))) __bf16 v16bf;
typedef __attribute__((ext_vector_type(8)))  float  v8f;
typedef __attribute__((ext_vector_type(4)))  int    v4i_t;

#define AS1 __attribute__((address_space(1)))
#define AS3 __attribute__((address_space(3)))

// CDNA5 data-movement feature detection (host pass sees none of these, so the
// fallback path keeps the TU parseable for x86).
#if defined(__HIP_DEVICE_COMPILE__) && __has_builtin(__builtin_amdgcn_global_load_async_to_lds_b128)
#define HAVE_ASYNC_LDS 1
#endif
#if defined(__HIP_DEVICE_COMPILE__) && __has_builtin(__builtin_amdgcn_tensor_load_to_lds)
#define HAVE_TDM 1
#endif

// Fragment assembly helper: two 16-byte chunks -> one 32-byte WMMA operand.
union Frag {
    uint4 u[2];
    v16bf v;
};

__device__ __forceinline__ unsigned short f2bf(float f) {
    unsigned int u = __float_as_uint(f);
    unsigned int r = (u + 0x7FFFu + ((u >> 16) & 1u)) >> 16;   // RNE
    return (unsigned short)r;
}
__device__ __forceinline__ float bf2f(unsigned short h) {
    return __uint_as_float(((unsigned int)h) << 16);
}

#if defined(HAVE_ASYNC_LDS)
__device__ __forceinline__ void async_copy_b128(const unsigned short* g,
                                                unsigned short* l) {
    // Builtin signature (from hipcc diagnostic): (v4i AS1*, v4i AS3*, Ii, Ii)
    __builtin_amdgcn_global_load_async_to_lds_b128(
        (AS1 v4i_t*)g, (AS3 v4i_t*)l, 0, 0);
}
__device__ __forceinline__ void wait_async0() {
#if __has_builtin(__builtin_amdgcn_s_wait_asynccnt)
    __builtin_amdgcn_s_wait_asynccnt(0);
#else
    asm volatile("s_wait_asynccnt 0" ::: "memory");
#endif
}
#endif

#if defined(HAVE_TDM)
// Issue one 2D TDM descriptor: tile_d1 rows of tile_d0 bf16 elements,
// row stride row_stride_elems, destination LDS with optional row padding.
__device__ __forceinline__ void tdm_load_2d_to_lds(
    unsigned lds_off_bytes, const void* gaddr,
    unsigned tile_d0, unsigned tile_d1,
    unsigned long long row_stride_elems,
    unsigned pad_interval_code, unsigned pad_amount_code, int pad_en) {
    typedef __attribute__((ext_vector_type(4))) unsigned int v4u_;
    typedef __attribute__((ext_vector_type(8))) int v8i_;
    typedef __attribute__((ext_vector_type(4))) int v4i_;

    unsigned long long ga = (unsigned long long)(uintptr_t)gaddr;

    v4u_ g0;
    g0[0] = 1u;                                           // count=1, user D#
    g0[1] = lds_off_bytes;                                // lds_addr
    g0[2] = (unsigned)(ga & 0xFFFFFFFFu);                 // global_addr lo
    g0[3] = (unsigned)((ga >> 32) & 0x01FFFFFFu)          // global_addr hi
          | (2u << 30);                                   // type = 2 (image)

    const unsigned d0 = (1u << 16)                        // data_size = 2B
                      | ((unsigned)(pad_en ? 1u : 0u) << 20)
                      | (pad_interval_code << 22)
                      | (pad_amount_code << 25);
    const unsigned td0 = tile_d0;                         // tensor_dim0
    const unsigned td1 = 0x00100000u;                     // generous dim1 bound

    v8i_ g1;
    g1[0] = (int)d0;
    g1[1] = (int)((td0 & 0xFFFFu) << 16);                 // dim0[15:0] @63:48
    g1[2] = (int)((td0 >> 16) | ((td1 & 0xFFFFu) << 16)); // dim0 hi | dim1 lo
    g1[3] = (int)((td1 >> 16) | ((tile_d0 & 0xFFFFu) << 16)); // tile_dim0 @127:112
    g1[4] = (int)(tile_d1 & 0xFFFFu);                     // tile_dim1; tile_dim2=0
    g1[5] = (int)(unsigned)(row_stride_elems & 0xFFFFFFFFull); // dim0_stride lo32
    g1[6] = (int)(unsigned)((row_stride_elems >> 32) & 0xFFFFull); // hi16
    g1[7] = 0;

    v4i_ gz = {0, 0, 0, 0};
#if __clang_major__ >= 23
    v8i_ gz8 = {0, 0, 0, 0, 0, 0, 0, 0};
    __builtin_amdgcn_tensor_load_to_lds(g0, g1, gz, gz, gz8, 0);
#else
    __builtin_amdgcn_tensor_load_to_lds(g0, g1, gz, gz, 0);
#endif
}
#endif

// ---------------------------------------------------------------------------
// Kernel 1: fp32 -> bf16 conversion (vectorized, grid-stride)
// ---------------------------------------------------------------------------
__global__ void cvt_f32_to_bf16(const float* __restrict__ src,
                                unsigned short* __restrict__ dst,
                                long long n) {
    long long i      = (long long)blockIdx.x * blockDim.x + threadIdx.x;
    long long stride = (long long)gridDim.x * blockDim.x;
    for (long long j = i * 4; j < n; j += stride * 4) {
        float4 f = *(const float4*)(src + j);
        ushort4 o;
        o.x = f2bf(f.x); o.y = f2bf(f.y); o.z = f2bf(f.z); o.w = f2bf(f.w);
        *(ushort4*)(dst + j) = o;
    }
}

// ---------------------------------------------------------------------------
// Kernel 2: NT GEMM, C[m,n] = sum_k A[m,k] * B[n,k]   (x @ W^T form)
// A: [M,K] bf16 row-major, B: [N,K] bf16 row-major.
// Block tile 128x128, K-chunk 32, 8 waves in 2(M) x 4(N), each wave 64x32
// (4x2 grid of 16x16 WMMA tiles). With async-to-LDS support the tiles are
// double-buffered and DMA'd memory->LDS (ASYNCcnt) with no VGPR staging.
// ---------------------------------------------------------------------------
template <bool OUT_F32>
__global__ __launch_bounds__(256, 1)
void gemm_nt_bf16(const unsigned short* __restrict__ A,
                  const unsigned short* __restrict__ Bw,
                  void* __restrict__ Cv,
                  int M, int N, int K) {
    constexpr int LDT = 40;                 // LDS row stride in halves (32 + 8)
#if defined(HAVE_ASYNC_LDS)
    __shared__ unsigned short lA[2][128 * LDT];
    __shared__ unsigned short lB[2][128 * LDT];
#else
    __shared__ unsigned short lA[1][128 * LDT];
    __shared__ unsigned short lB[1][128 * LDT];
#endif

    const int tid  = threadIdx.x;
    const int lane = tid & 31;
    const int wave = tid >> 5;
    const int wm   = wave >> 2;             // 0..1 : 64 rows each
    const int wn   = wave & 3;              // 0..3 : 32 cols each
    const int l16  = lane & 15;
    const int hs   = lane >> 4;             // half-wave select

    const int rowBase = blockIdx.y * 128;
    const int colBase = blockIdx.x * 128;

    v8f acc[4][2] = {};

    const int idx0 = tid, idx1 = tid + 256;
    const int r0 = idx0 >> 2, c0 = idx0 & 3;   // 4x uint4 (32 halves) per row
    const int r1 = idx1 >> 2, c1 = idx1 & 3;

    // WMMA consumption of one staged K-chunk (ISA bf16 A 16x32 / B 32x16
    // lane layouts; two 16B ds_load_b128 per fragment).
    auto compute_chunk = [&](const unsigned short* tA, const unsigned short* tB) {
        Frag af[4];
#pragma unroll
        for (int i = 0; i < 4; ++i) {
            const int row = wm * 64 + i * 16 + l16;
            const unsigned short* p = &tA[row * LDT + hs * 8];
            af[i].u[0] = *(const uint4*)(p);
            af[i].u[1] = *(const uint4*)(p + 16);   // K += 16 halves
        }
        Frag bf_[2];
#pragma unroll
        for (int j = 0; j < 2; ++j) {
            const int col = wn * 32 + j * 16 + l16;
            const unsigned short* p = &tB[col * LDT + hs * 8];
            bf_[j].u[0] = *(const uint4*)(p);
            bf_[j].u[1] = *(const uint4*)(p + 16);
        }
#pragma unroll
        for (int i = 0; i < 4; ++i)
#pragma unroll
            for (int j = 0; j < 2; ++j)
                acc[i][j] = __builtin_amdgcn_wmma_f32_16x16x32_bf16(
                    false, af[i].v, false, bf_[j].v, (short)0, acc[i][j],
                    false, false);
    };

#if defined(HAVE_ASYNC_LDS)
    // ---- async double-buffered pipeline: DMA tile kk+32 while computing kk --
    auto issue_tile = [&](int kk, int buf) {
        async_copy_b128(A  + (long long)(rowBase + r0) * K + kk + c0 * 8,
                        &lA[buf][r0 * LDT + c0 * 8]);
        async_copy_b128(A  + (long long)(rowBase + r1) * K + kk + c1 * 8,
                        &lA[buf][r1 * LDT + c1 * 8]);
        async_copy_b128(Bw + (long long)(colBase + r0) * K + kk + c0 * 8,
                        &lB[buf][r0 * LDT + c0 * 8]);
        async_copy_b128(Bw + (long long)(colBase + r1) * K + kk + c1 * 8,
                        &lB[buf][r1 * LDT + c1 * 8]);
    };
    issue_tile(0, 0);
    wait_async0();
    __syncthreads();
    for (int kk = 0; kk < K; kk += 32) {
        const int cur = (kk >> 5) & 1;
        if (kk + 32 < K) issue_tile(kk + 32, cur ^ 1);   // prefetch next buffer
        compute_chunk(lA[cur], lB[cur]);
        wait_async0();                                   // next tile landed
        __syncthreads();                                 // everyone done w/ cur
    }
#else
    // ---- fallback: VGPR staging, single buffer ----
    for (int kk = 0; kk < K; kk += 32) {
        uint4 a0 = *(const uint4*)(A  + (long long)(rowBase + r0) * K + kk + c0 * 8);
        uint4 a1 = *(const uint4*)(A  + (long long)(rowBase + r1) * K + kk + c1 * 8);
        uint4 b0 = *(const uint4*)(Bw + (long long)(colBase + r0) * K + kk + c0 * 8);
        uint4 b1 = *(const uint4*)(Bw + (long long)(colBase + r1) * K + kk + c1 * 8);
        __builtin_prefetch(A  + (long long)(rowBase + r0) * K + kk + 32, 0, 0);
        __builtin_prefetch(Bw + (long long)(colBase + r0) * K + kk + 32, 0, 0);

        __syncthreads();
        *(uint4*)&lA[0][r0 * LDT + c0 * 8] = a0;
        *(uint4*)&lA[0][r1 * LDT + c1 * 8] = a1;
        *(uint4*)&lB[0][r0 * LDT + c0 * 8] = b0;
        *(uint4*)&lB[0][r1 * LDT + c1 * 8] = b1;
        __syncthreads();

        compute_chunk(lA[0], lB[0]);
    }
#endif

    // ---- epilogue: C layout lane(col)=l16, VGPR r -> row r (+8 for hs=1) ----
#pragma unroll
    for (int i = 0; i < 4; ++i) {
#pragma unroll
        for (int j = 0; j < 2; ++j) {
            const int col   = colBase + wn * 32 + j * 16 + l16;
            const int rbase = rowBase + wm * 64 + i * 16 + hs * 8;
#pragma unroll
            for (int r = 0; r < 8; ++r) {
                const float v = acc[i][j][r];
                if (OUT_F32)
                    ((float*)Cv)[(long long)(rbase + r) * N + col] = v;
                else
                    ((unsigned short*)Cv)[(long long)(rbase + r) * N + col] = f2bf(v);
            }
        }
    }
}

// ---------------------------------------------------------------------------
// Kernel 3: in-place Llama RoPE on bf16 tensor [B,S,nheads,128].
// ---------------------------------------------------------------------------
__global__ void rope_kernel(unsigned short* __restrict__ X, int nheads) {
    long long idx   = (long long)blockIdx.x * blockDim.x + threadIdx.x;
    long long total = (long long)BATCH * SEQ * nheads * 64;
    if (idx >= total) return;
    const int  i    = (int)(idx & 63);
    long long  t    = idx >> 6;                 // (b*SEQ + s)*nheads + h
    const int  srow = (int)((t / nheads) % SEQ);
    long long  base = t * (long long)HEAD_DIM;

    const float inv = __powf(10000.0f, -(2.0f * (float)i) / 128.0f);
    const float ang = (float)srow * inv;
    float sn, cs;
    __sincosf(ang, &sn, &cs);

    const float x1 = bf2f(X[base + i]);
    const float x2 = bf2f(X[base + i + 64]);
    X[base + i]      = f2bf(x1 * cs - x2 * sn);
    X[base + i + 64] = f2bf(x2 * cs + x1 * sn);
}

// ---------------------------------------------------------------------------
// Kernel 4: flash attention (causal, GQA rep=4), bf16 WMMA, f32 accum.
// Grid: (SEQ/128, NUM_HEADS, BATCH); block 256 = 8 waves, each wave owns
// 16 query rows. K tile is fetched by the Tensor Data Mover (wave 0 issues a
// 2D D# whose pad fields reproduce the LDS row padding; TENSORcnt + barrier
// publish it). V is staged manually because it must be transposed into LDS.
// ---------------------------------------------------------------------------
__global__ __launch_bounds__(256, 1)
void flash_attn_kernel(const unsigned short* __restrict__ Q,   // [B,S,H,D]
                       const unsigned short* __restrict__ Kg,  // [B,S,KVH,D]
                       const unsigned short* __restrict__ Vg,  // [B,S,KVH,D]
                       unsigned short* __restrict__ Og) {      // [B,S,H,D]
    constexpr int BK   = 64;     // keys per tile
    constexpr int KSTR = 136;    // lK row stride (128 D + 8 pad = 64+4 DWORDs)
    constexpr int VSTR = 72;     // lVt row stride (64 keys + 8 pad)
    constexpr int PSTR = 72;     // lP row stride
    __shared__ unsigned short lK[BK * KSTR];          // [key][d]
    __shared__ unsigned short lVt[HEAD_DIM * VSTR];   // [d][key] (transposed)
    __shared__ unsigned short lP[8 * 16 * PSTR];      // per-wave P bounce

    const int tid  = threadIdx.x;
    const int lane = tid & 31;
    const int wave = tid >> 5;
    const int l16  = lane & 15;
    const int hs   = lane >> 4;

    const int qblk  = blockIdx.x;
    const int h     = blockIdx.y;
    const int b     = blockIdx.z;
    const int kvh   = h >> 2;                 // GQA: 4 query heads per KV head
    const int qbase = qblk * 128;
    const int qw    = qbase + wave * 16;      // this wave's first query row

    // ---- load Q fragments once (A-matrix 16x32 layout, direct from global) --
    Frag qf[4];
    {
        const unsigned short* qptr =
            Q + (((long long)b * SEQ + qw + l16) * NUM_HEADS + h) * HEAD_DIM;
#pragma unroll
        for (int c = 0; c < 4; ++c) {
            qf[c].u[0] = *(const uint4*)(qptr + c * 32 + hs * 8);
            qf[c].u[1] = *(const uint4*)(qptr + c * 32 + hs * 8 + 16);
        }
    }

    float m_r[8], l_r[8];
    v8f   o_acc[8] = {};
#pragma unroll
    for (int r = 0; r < 8; ++r) { m_r[r] = -3.0e38f; l_r[r] = 0.0f; }

    const float scale = 0.08838834764831845f;   // 1/sqrt(128)
    const int ntiles  = (qbase + 128) / BK;     // causal: keys <= qbase+127

    for (int t = 0; t < ntiles; ++t) {
        const int kb = t * BK;
        const long long kvRowBase = ((long long)b * SEQ + kb) * NUM_KV_HEADS + kvh;

        // ---- stage transposed V tile through registers ----
        uint4 vreg[4];
#pragma unroll
        for (int it = 0; it < 4; ++it) {
            const int idx = tid + it * 256;     // 1024 uint4 per tensor
            const int r   = idx >> 4;           // key row (16 uint4 per row)
            const int c   = idx & 15;
            vreg[it] = *(const uint4*)(Vg + (kvRowBase + (long long)r * NUM_KV_HEADS)
                                                * HEAD_DIM + c * 8);
        }
#if !defined(HAVE_TDM)
        uint4 kreg[4];
#pragma unroll
        for (int it = 0; it < 4; ++it) {
            const int idx = tid + it * 256;
            const int r   = idx >> 4;
            const int c   = idx & 15;
            kreg[it] = *(const uint4*)(Kg + (kvRowBase + (long long)r * NUM_KV_HEADS)
                                                * HEAD_DIM + c * 8);
        }
#endif
        __syncthreads();                        // prev iter done reading LDS

#if defined(HAVE_TDM)
        if (wave == 0) {
            const unsigned lds_off =
                (unsigned)(uintptr_t)(AS3 void*)&lK[0];
            const unsigned short* gk = Kg + kvRowBase * HEAD_DIM;
            // 64 rows x 128 bf16, row stride KVH*D elems; LDS pad: every
            // 64 DWORDs (code 5) insert 4 DWORDs (code 3) -> KSTR=136 halves.
            tdm_load_2d_to_lds(lds_off, gk, HEAD_DIM, BK,
                               (unsigned long long)(NUM_KV_HEADS * HEAD_DIM),
                               5u, 3u, 1);
        }
#else
#pragma unroll
        for (int it = 0; it < 4; ++it) {
            const int idx = tid + it * 256;
            const int r   = idx >> 4;
            const int c   = idx & 15;
            *(uint4*)&lK[r * KSTR + c * 8] = kreg[it];
        }
#endif
#pragma unroll
        for (int it = 0; it < 4; ++it) {
            const int idx = tid + it * 256;
            const int r   = idx >> 4;
            const int c   = idx & 15;
            unsigned short tmp[8];
            *(uint4*)tmp = vreg[it];
#pragma unroll
            for (int e = 0; e < 8; ++e)           // transpose V: [d][key]
                lVt[(c * 8 + e) * VSTR + r] = tmp[e];
        }
#if defined(HAVE_TDM)
        if (wave == 0) __builtin_amdgcn_s_wait_tensorcnt(0);
#endif
        __syncthreads();                        // K (TDM) + V (ds) published

        // ---- S = Q K^T : 4 key-subtiles x 4 d-chunks of WMMA ----
        v8f sacc[4] = {};
#pragma unroll
        for (int n = 0; n < 4; ++n) {
            const int key = n * 16 + l16;
#pragma unroll
            for (int c = 0; c < 4; ++c) {
                Frag kf;
                const unsigned short* p = &lK[key * KSTR + c * 32 + hs * 8];
                kf.u[0] = *(const uint4*)(p);
                kf.u[1] = *(const uint4*)(p + 16);
                sacc[n] = __builtin_amdgcn_wmma_f32_16x16x32_bf16(
                    false, qf[c].v, false, kf.v, (short)0, sacc[n], false, false);
            }
        }

        // ---- scale + causal mask ----
        const int myq = qw + hs * 8;           // + r gives this lane's row
#pragma unroll
        for (int n = 0; n < 4; ++n) {
            const int key = kb + n * 16 + l16;
#pragma unroll
            for (int r = 0; r < 8; ++r) {
                float s = sacc[n][r] * scale;
                if (key > myq + r) s = -3.0e38f;
                sacc[n][r] = s;
            }
        }

        // ---- online softmax (row = 16 lanes of a half-wave) ----
#pragma unroll
        for (int r = 0; r < 8; ++r) {
            float tmax = -3.0e38f;
#pragma unroll
            for (int n = 0; n < 4; ++n) tmax = fmaxf(tmax, sacc[n][r]);
            for (int off = 1; off < 16; off <<= 1)
                tmax = fmaxf(tmax, __shfl_xor(tmax, off, 32));
            const float mnew = fmaxf(m_r[r], tmax);
            const float corr = __expf(m_r[r] - mnew);
            m_r[r] = mnew;
            float rsum = 0.0f;
#pragma unroll
            for (int n = 0; n < 4; ++n) {
                const float p = __expf(sacc[n][r] - mnew);
                sacc[n][r] = p;
                rsum += p;
            }
            for (int off = 1; off < 16; off <<= 1)
                rsum += __shfl_xor(rsum, off, 32);
            l_r[r] = l_r[r] * corr + rsum;
#pragma unroll
            for (int nn = 0; nn < 8; ++nn) o_acc[nn][r] *= corr;
        }

        // ---- P: C-layout -> A-layout via per-wave LDS bounce (in-order DS) --
        unsigned short* pw = &lP[wave * 16 * PSTR];
#pragma unroll
        for (int n = 0; n < 4; ++n)
#pragma unroll
            for (int r = 0; r < 8; ++r)
                pw[(hs * 8 + r) * PSTR + n * 16 + l16] = f2bf(sacc[n][r]);

        // ---- O += P V : 2 key-chunks x 8 d-subtiles of WMMA ----
#pragma unroll
        for (int c = 0; c < 2; ++c) {
            Frag pf;
            const unsigned short* pp = &pw[l16 * PSTR + c * 32 + hs * 8];
            pf.u[0] = *(const uint4*)(pp);
            pf.u[1] = *(const uint4*)(pp + 16);
#pragma unroll
            for (int nn = 0; nn < 8; ++nn) {
                Frag vf;
                const int d = nn * 16 + l16;
                const unsigned short* vp = &lVt[d * VSTR + c * 32 + hs * 8];
                vf.u[0] = *(const uint4*)(vp);
                vf.u[1] = *(const uint4*)(vp + 16);
                o_acc[nn] = __builtin_amdgcn_wmma_f32_16x16x32_bf16(
                    false, pf.v, false, vf.v, (short)0, o_acc[nn], false, false);
            }
        }
    }

    // ---- normalize and store attn output bf16 [B,S,H,D] ----
    unsigned short* op =
        Og + (((long long)b * SEQ + qw) * NUM_HEADS + h) * HEAD_DIM;
#pragma unroll
    for (int nn = 0; nn < 8; ++nn) {
        const int d = nn * 16 + l16;
#pragma unroll
        for (int r = 0; r < 8; ++r) {
            const float v = o_acc[nn][r] * (1.0f / l_r[r]);
            op[(long long)(hs * 8 + r) * (NUM_HEADS * HEAD_DIM) + d] = f2bf(v);
        }
    }
}

// ---------------------------------------------------------------------------
// Host-side orchestration
// ---------------------------------------------------------------------------
extern "C" void kernel_launch(void* const* d_in, const int* in_sizes, int n_in,
                              void* d_out, int out_size, void* d_ws, size_t ws_size,
                              hipStream_t stream) {
    (void)in_sizes; (void)n_in; (void)out_size; (void)ws_size;

    const float* X  = (const float*)d_in[0];
    const float* Wq = (const float*)d_in[1];
    const float* Wk = (const float*)d_in[2];
    const float* Wv = (const float*)d_in[3];
    const float* Wo = (const float*)d_in[4];
    float* out = (float*)d_out;

    const long long nX  = (long long)BATCH * SEQ * HIDDEN;      // 16.7M
    const long long nWq = (long long)HIDDEN * HIDDEN;           // 16.7M
    const long long nWk = (long long)KV_HIDDEN * HIDDEN;        // 4.2M
    const long long nKV = (long long)BATCH * SEQ * KV_HIDDEN;   // 4.2M

    unsigned short* ws  = (unsigned short*)d_ws;
    unsigned short* Xb  = ws;  ws += nX;
    unsigned short* Wqb = ws;  ws += nWq;
    unsigned short* Wkb = ws;  ws += nWk;
    unsigned short* Wvb = ws;  ws += nWk;
    unsigned short* Wob = ws;  ws += nWq;
    unsigned short* Qb  = ws;  ws += nX;
    unsigned short* Kb  = ws;  ws += nKV;
    unsigned short* Vb  = ws;  ws += nKV;
    unsigned short* Ab  = ws;  ws += nX;

    // 1) bf16 conversion (compute-bound problem; one cheap pass over ~300MB)
    auto cvt = [&](const float* s, unsigned short* d, long long n) {
        int blocks = (int)((n / 4 + 255) / 256);
        if (blocks > 4096) blocks = 4096;
        cvt_f32_to_bf16<<<dim3(blocks), dim3(256), 0, stream>>>(s, d, n);
    };
    cvt(X,  Xb,  nX);
    cvt(Wq, Wqb, nWq);
    cvt(Wk, Wkb, nWk);
    cvt(Wv, Wvb, nWk);
    cvt(Wo, Wob, nWq);

    const int Mrows = BATCH * SEQ;   // 4096

    // 2) QKV projections (x @ W^T), bf16 out
    gemm_nt_bf16<false><<<dim3(HIDDEN / 128,    Mrows / 128), 256, 0, stream>>>(
        Xb, Wqb, Qb, Mrows, HIDDEN, HIDDEN);
    gemm_nt_bf16<false><<<dim3(KV_HIDDEN / 128, Mrows / 128), 256, 0, stream>>>(
        Xb, Wkb, Kb, Mrows, KV_HIDDEN, HIDDEN);
    gemm_nt_bf16<false><<<dim3(KV_HIDDEN / 128, Mrows / 128), 256, 0, stream>>>(
        Xb, Wvb, Vb, Mrows, KV_HIDDEN, HIDDEN);

    // 3) RoPE on Q and K
    {
        long long tq = (long long)BATCH * SEQ * NUM_HEADS * 64;
        rope_kernel<<<dim3((int)((tq + 255) / 256)), 256, 0, stream>>>(Qb, NUM_HEADS);
        long long tk = (long long)BATCH * SEQ * NUM_KV_HEADS * 64;
        rope_kernel<<<dim3((int)((tk + 255) / 256)), 256, 0, stream>>>(Kb, NUM_KV_HEADS);
    }

    // 4) causal GQA flash attention -> Ab [B,S,H*D] bf16
    flash_attn_kernel<<<dim3(SEQ / 128, NUM_HEADS, BATCH), 256, 0, stream>>>(
        Qb, Kb, Vb, Ab);

    // 5) output projection, fp32 straight into d_out
    gemm_nt_bf16<true><<<dim3(HIDDEN / 128, Mrows / 128), 256, 0, stream>>>(
        Ab, Wob, out, Mrows, HIDDEN, HIDDEN);
}
